// RnnDecoder_15590731284673
// MI455X (gfx1250) — matrix operand order repaired and checked by reading, not compiled, not run on hardware
//
#include <hip/hip_runtime.h>
#include <hip/hip_bf16.h>

#define L_ 128
#define B_ 128
#define A_ 512
#define I_ 512
#define H_ 1024
#define G_ 4096  // 4*H

typedef __attribute__((ext_vector_type(16))) _Float16 v16h;
typedef __attribute__((ext_vector_type(8)))  _Float16 v8h;
typedef __attribute__((ext_vector_type(8)))  float    v8f;

// ---------------------------------------------------------------------------
// LDS helpers: generic -> LDS(as3) byte offset for async-to-LDS instructions
// ---------------------------------------------------------------------------
typedef __attribute__((address_space(3))) void lds_void_t;
__device__ __forceinline__ unsigned lds_addr(void* p) {
  return (unsigned)(unsigned long long)(lds_void_t*)p;
}

// GLOBAL_LOAD_ASYNC_TO_LDS_B128: per-lane 16B global -> LDS, tracked by ASYNCcnt
__device__ __forceinline__ void async_ld_b128(unsigned lds_off, const void* g) {
  asm volatile("global_load_async_to_lds_b128 %0, %1, off"
               :: "v"(lds_off), "v"(g)
               : "memory");
}

#define WAIT_ASYNC(n)                                              \
  do {                                                             \
    asm volatile("s_wait_asynccnt " #n ::: "memory");              \
  } while (0)

// ---------------------------------------------------------------------------
// Fragment helpers (WMMA 16-bit operands in this code's contiguous k order, not the ISA table order)
// A (16x32, MxK): lanes 0-15 -> M=lane, K = k0+{0..15};
//                 lanes 16-31 -> M=lane-16, K = k0+{16..31}
// B (32x16, KxN): pre-packed so lane L holds halves j=0..15 with
//                 K = k0 + ((L>=16)?16:0) + j, N = n0 + (L&15)
// ---------------------------------------------------------------------------
__device__ __forceinline__ v16h load_a16(const _Float16* __restrict__ row,
                                         int kbase, int lane) {
  const int off = kbase + ((lane & 16) ? 16 : 0);
  v8h lo = *(const v8h*)(row + off);
  v8h hi = *(const v8h*)(row + off + 8);
  v16h a;
#pragma unroll
  for (int t = 0; t < 8; ++t) { a[t] = lo[t]; a[8 + t] = hi[t]; }
  return a;
}

// ---------------------------------------------------------------------------
// One-time packing / conversion kernels
// ---------------------------------------------------------------------------
__global__ __launch_bounds__(256) void pack_wcat_kernel(
    const float* __restrict__ Wih, const float* __restrict__ Whh,
    _Float16* __restrict__ out) {
  const int numK = (A_ + H_) / 32;  // 48
  size_t o = (size_t)blockIdx.x * 256 + threadIdx.x;
  size_t tile = o >> 9;  // 512 halves per 32x16 tile
  int within = (int)(o & 511);
  int lane = within >> 4, j = within & 15;
  int ntile = (int)(tile / numK), kc = (int)(tile % numK);
  int n = ntile * 16 + (lane & 15);
  int k = kc * 32 + ((lane >= 16) ? 16 : 0) + j;
  float v = (k < A_) ? Wih[(size_t)n * (I_ + A_) + I_ + k]
                     : Whh[(size_t)n * H_ + (k - A_)];
  out[o] = (_Float16)v;
}

__global__ __launch_bounds__(256) void pack_win_kernel(
    const float* __restrict__ Wih, _Float16* __restrict__ out) {
  const int numK = I_ / 32;  // 16
  size_t o = (size_t)blockIdx.x * 256 + threadIdx.x;
  size_t tile = o >> 9;
  int within = (int)(o & 511);
  int lane = within >> 4, j = within & 15;
  int ntile = (int)(tile / numK), kc = (int)(tile % numK);
  int n = ntile * 16 + (lane & 15);
  int k = kc * 32 + ((lane >= 16) ? 16 : 0) + j;
  out[o] = (_Float16)Wih[(size_t)n * (I_ + A_) + k];
}

__global__ __launch_bounds__(256) void cvt_f16_kernel(
    const float* __restrict__ in, _Float16* __restrict__ out, int n) {
  int i = blockIdx.x * 256 + threadIdx.x;
  if (i < n) out[i] = (_Float16)in[i];
}

__global__ __launch_bounds__(256) void init_state_kernel(
    const float* __restrict__ h0, const float* __restrict__ c0,
    const float* __restrict__ bih, const float* __restrict__ bhh,
    _Float16* __restrict__ h16, float* __restrict__ cbuf,
    float* __restrict__ biassum) {
  int i = blockIdx.x * 256 + threadIdx.x;
  if (i < G_) biassum[i] = bih[i] + bhh[i];
  if (i < B_ * H_) {
    h16[i] = (_Float16)h0[i];
    cbuf[i] = c0[i];
  }
}

// ---------------------------------------------------------------------------
// base = input_feature @ W_ih[:, :I]^T + (b_ih + b_hh)    [B, 4H]
// grid: (G_/128, B_/16), block: 256 (8 waves, one 16x16 tile each)
// ---------------------------------------------------------------------------
__global__ __launch_bounds__(256) void base_gemm_kernel(
    const _Float16* __restrict__ in16, const _Float16* __restrict__ win,
    const float* __restrict__ biassum, float* __restrict__ basebuf) {
  const int lane = threadIdx.x & 31;
  const int w = threadIdx.x >> 5;
  const int m0 = blockIdx.y * 16;
  const int n0 = blockIdx.x * 128 + w * 16;
  const int numK = I_ / 32;  // 16
  const int ntile = n0 >> 4;

  v8f c;
  const float bv = biassum[n0 + (lane & 15)];
#pragma unroll
  for (int r = 0; r < 8; ++r) c[r] = bv;

  const _Float16* arow = in16 + (size_t)(m0 + (lane & 15)) * I_;
  const _Float16* wtile = win + ((size_t)(ntile * numK) << 9);
  for (int kc = 0; kc < numK; ++kc) {
    v16h a = load_a16(arow, kc * 32, lane);
    v16h b = *(const v16h*)(wtile + ((size_t)kc << 9) + (lane << 4));
    if (kc + 1 < numK)
      __builtin_prefetch(wtile + ((size_t)(kc + 1) << 9) + (lane << 4), 0, 1);
    c = __builtin_amdgcn_wmma_f32_16x16x32_f16(false, a, false, b, (short)0,
                                               c, false, false);
  }
  const int rowadd = (lane & 16) ? 8 : 0;
  const int col = n0 + (lane & 15);
#pragma unroll
  for (int r = 0; r < 8; ++r)
    basebuf[(size_t)(m0 + rowadd + r) * G_ + col] = c[r];
}

// ---------------------------------------------------------------------------
// One LSTM step, fused: gates = base + [a_l, h] @ [Wa; Wh]^T, then activation.
//
// Block macro-tile: 64 batch rows x 16 hidden cols x all 4 gates.
//   grid: (H_/16 = 64, B_/64 = 2), block 256 = 8 waves.
//   wave w: row-tile rt = w&3 (16 rows), gate pair gp = w>>2 -> gates {2gp,2gp+1}
//   -> one A fragment feeds two WMMAs (in-register A reuse).
// A (64x32 f16 chunk) and B (4 pre-swizzled gate tiles) are staged to LDS via
// GLOBAL_LOAD_ASYNC_TO_LDS_B128, double-buffered on ASYNCcnt.
// ---------------------------------------------------------------------------
__global__ __launch_bounds__(256) void lstm_step_kernel(
    int l, const _Float16* __restrict__ act16,
    const _Float16* __restrict__ h16_in, _Float16* __restrict__ h16_out,
    const float* __restrict__ c_in, float* __restrict__ c_out,
    const _Float16* __restrict__ wcat, const float* __restrict__ basebuf,
    float* __restrict__ hs_out) {
  const int numK = (A_ + H_) / 32;  // 48
  const int AROW = 40;              // padded A row stride (halves): 80B -> no bank conflicts

  __shared__ _Float16 sA[2][64 * AROW];  // 10.0 KB
  __shared__ _Float16 sB[2][4 * 512];    //  8.0 KB
  __shared__ float gsm[4][64][16];       // 16.0 KB

  const int tid = threadIdx.x;
  const int lane = tid & 31;
  const int w = tid >> 5;
  const int rt = w & 3;        // row tile within block
  const int gp = w >> 2;       // gate pair: 0 -> {0,1}, 1 -> {2,3}
  const int g0 = gp * 2, g1 = gp * 2 + 1;
  const int mblk = blockIdx.y * 64;
  const int hcol0 = blockIdx.x * 16;
  const int coll = lane & 15;
  const int rowadd = (lane & 16) ? 8 : 0;

  const unsigned sA_base = lds_addr(&sA[0][0]);
  const unsigned sB_base = lds_addr(&sB[0][0]);

  // ---- async stage of K-chunk kc into buffer buf ----
  auto stage = [&](int kc, int buf) {
    // A: 64 rows x 32 halves; thread t moves 16B: row = t>>2, seg = t&3
    {
      const int row = tid >> 2, seg = tid & 3;
      const _Float16* src =
          (kc < A_ / 32)
              ? act16 + ((size_t)l * B_ + mblk + row) * A_ + kc * 32 + seg * 8
              : h16_in + (size_t)(mblk + row) * H_ + (kc - A_ / 32) * 32 +
                    seg * 8;
      async_ld_b128(sA_base + (unsigned)(buf * 64 * AROW + row * AROW + seg * 8) * 2,
                    src);
    }
    // B: 4 gate tiles x 512 halves; thread t moves 16B: gate = t>>6
    {
      const int gate = tid >> 6, within = tid & 63;
      const int ntile = gate * (H_ / 16) + blockIdx.x;  // col tile in gate space
      const _Float16* src =
          wcat + (((size_t)ntile * numK + kc) << 9) + within * 8;
      async_ld_b128(sB_base + (unsigned)(buf * 2048 + gate * 512 + within * 8) * 2,
                    src);
    }
  };

  // ---- seed accumulators with base (input projection + biases) ----
  const int m0 = mblk + rt * 16;
  v8f acc0, acc1;
#pragma unroll
  for (int r = 0; r < 8; ++r) {
    size_t rowoff = (size_t)(m0 + rowadd + r) * G_;
    acc0[r] = basebuf[rowoff + g0 * H_ + hcol0 + coll];
    acc1[r] = basebuf[rowoff + g1 * H_ + hcol0 + coll];
  }

  // ---- pipelined K loop: stage ahead with async-to-LDS, compute with WMMA ----
  stage(0, 0);
  stage(1, 1);
  WAIT_ASYNC(0x2);  // chunk 0 resident (chunk 1 may be in flight)
  __syncthreads();

  for (int kc = 0; kc < numK; ++kc) {
    const int buf = kc & 1;
    // A fragment from LDS (shared by both gate tiles of this wave)
    const _Float16* abase =
        &sA[buf][(rt * 16 + coll) * AROW + ((lane & 16) ? 16 : 0)];
    v8h lo = *(const v8h*)abase;
    v8h hi = *(const v8h*)(abase + 8);
    v16h a;
#pragma unroll
    for (int t = 0; t < 8; ++t) { a[t] = lo[t]; a[8 + t] = hi[t]; }
    v16h b0 = *(const v16h*)&sB[buf][g0 * 512 + lane * 16];
    v16h b1 = *(const v16h*)&sB[buf][g1 * 512 + lane * 16];
    acc0 = __builtin_amdgcn_wmma_f32_16x16x32_f16(false, a, false, b0, (short)0,
                                                  acc0, false, false);
    acc1 = __builtin_amdgcn_wmma_f32_16x16x32_f16(false, a, false, b1, (short)0,
                                                  acc1, false, false);

    __syncthreads();  // everyone done reading buf before it is overwritten
    if (kc + 2 < numK) stage(kc + 2, buf);
    if (kc + 1 < numK) {
      if (kc + 2 < numK) {
        WAIT_ASYNC(0x2);  // chunk kc+1 resident; kc+2 still in flight
      } else {
        WAIT_ASYNC(0x0);  // last staged chunk resident
      }
      __syncthreads();
    }
  }

  // ---- park gate tiles in LDS for the fused activation ----
#pragma unroll
  for (int r = 0; r < 8; ++r) {
    gsm[g0][rt * 16 + rowadd + r][coll] = acc0[r];
    gsm[g1][rt * 16 + rowadd + r][coll] = acc1[r];
  }
  __syncthreads();

  // ---- elementwise LSTM cell: 64x16 = 1024 outputs, 4 per thread ----
#pragma unroll
  for (int s = 0; s < 4; ++s) {
    int idx = tid + s * 256;
    int row = idx >> 4;
    int col = idx & 15;
    float ig = gsm[0][row][col];
    float fg = gsm[1][row][col];
    float gg = gsm[2][row][col];
    float og = gsm[3][row][col];
    size_t p = (size_t)(mblk + row) * H_ + hcol0 + col;
    float cold = c_in[p];
    float si = 1.f / (1.f + __expf(-ig));
    float sf = 1.f / (1.f + __expf(-fg));
    float so = 1.f / (1.f + __expf(-og));
    float cnew = sf * cold + si * tanhf(gg);
    float h = so * tanhf(cnew);
    c_out[p] = cnew;
    h16_out[p] = (_Float16)h;
    hs_out[(size_t)l * B_ * H_ + p] = h;
  }
}

__global__ __launch_bounds__(256) void finalize_kernel(
    const float* __restrict__ hs_last, const float* __restrict__ cfin,
    float* __restrict__ hn, float* __restrict__ cn) {
  int i = blockIdx.x * 256 + threadIdx.x;
  if (i < B_ * H_) {
    hn[i] = hs_last[i];
    cn[i] = cfin[i];
  }
}

// ---------------------------------------------------------------------------
extern "C" void kernel_launch(void* const* d_in, const int* in_sizes, int n_in,
                              void* d_out, int out_size, void* d_ws,
                              size_t ws_size, hipStream_t stream) {
  const float* act   = (const float*)d_in[0];  // [L,B,A]
  const float* input = (const float*)d_in[1];  // [B,I]
  const float* h0    = (const float*)d_in[2];  // [B,H]
  const float* c0    = (const float*)d_in[3];  // [B,H]
  const float* Wih   = (const float*)d_in[4];  // [4H, I+A]
  const float* Whh   = (const float*)d_in[5];  // [4H, H]
  const float* bih   = (const float*)d_in[6];  // [4H]
  const float* bhh   = (const float*)d_in[7];  // [4H]

  char* ws = (char*)d_ws;
  size_t off = 0;
  auto alloc = [&](size_t bytes) {
    char* p = ws + off;
    off += (bytes + 255) & ~(size_t)255;
    return p;
  };
  _Float16* wcat  = (_Float16*)alloc((size_t)G_ * (A_ + H_) * 2);  // 12.6 MB
  _Float16* win   = (_Float16*)alloc((size_t)G_ * I_ * 2);         //  4.2 MB
  _Float16* act16 = (_Float16*)alloc((size_t)L_ * B_ * A_ * 2);    // 16.8 MB
  _Float16* in16  = (_Float16*)alloc((size_t)B_ * I_ * 2);
  float* biassum  = (float*)alloc((size_t)G_ * 4);
  float* basebuf  = (float*)alloc((size_t)B_ * G_ * 4);            //  2.1 MB
  _Float16* h16[2] = {(_Float16*)alloc((size_t)B_ * H_ * 2),
                      (_Float16*)alloc((size_t)B_ * H_ * 2)};
  float* cbuf[2] = {(float*)alloc((size_t)B_ * H_ * 4),
                    (float*)alloc((size_t)B_ * H_ * 4)};

  float* hs = (float*)d_out;                       // [L,B,H]
  float* hn = hs + (size_t)L_ * B_ * H_;           // [B,H]
  float* cn = hn + (size_t)B_ * H_;                // [B,H]

  pack_wcat_kernel<<<(G_ * (A_ + H_)) / 256, 256, 0, stream>>>(Wih, Whh, wcat);
  pack_win_kernel<<<(G_ * I_) / 256, 256, 0, stream>>>(Wih, win);
  cvt_f16_kernel<<<(L_ * B_ * A_) / 256, 256, 0, stream>>>(act, act16,
                                                           L_ * B_ * A_);
  cvt_f16_kernel<<<(B_ * I_) / 256, 256, 0, stream>>>(input, in16, B_ * I_);
  init_state_kernel<<<(B_ * H_) / 256, 256, 0, stream>>>(h0, c0, bih, bhh,
                                                         h16[0], cbuf[0],
                                                         biassum);
  base_gemm_kernel<<<dim3(G_ / 128, B_ / 16), 256, 0, stream>>>(in16, win,
                                                                biassum,
                                                                basebuf);
  for (int l = 0; l < L_; ++l) {
    lstm_step_kernel<<<dim3(H_ / 16, B_ / 64), 256, 0, stream>>>(
        l, act16, h16[l & 1], h16[(l + 1) & 1], cbuf[l & 1], cbuf[(l + 1) & 1],
        wcat, basebuf, hs);
  }
  finalize_kernel<<<(B_ * H_) / 256, 256, 0, stream>>>(
      hs + (size_t)(L_ - 1) * B_ * H_, cbuf[0], hn, cn);
}
